// BloodhoundSub_12463995093069
// MI455X (gfx1250) — matrix operand order repaired and checked by reading, not compiled, not run on hardware
//
#include <hip/hip_runtime.h>
#include <hip/hip_bf16.h>

// ---------------------------------------------------------------------------
// CDNA5 (gfx1250, wave32) transformer forward using v_wmma_f32_16x16x32_bf16.
// Data path: coalesced b128 global loads -> double-buffered LDS (bf16)
//            -> batched ds_load_b128 fragments -> back-to-back WMMA groups.
// ---------------------------------------------------------------------------

typedef __attribute__((ext_vector_type(16))) __bf16 v16bf;
typedef __attribute__((ext_vector_type(8)))  __bf16 v8bf;
typedef __attribute__((ext_vector_type(8)))  float  v8f;

__device__ __forceinline__ v8f wmma_bf16(const v16bf& a, const v16bf& b, const v8f& c) {
  return __builtin_amdgcn_wmma_f32_16x16x32_bf16(false, a, false, b, (short)0, c, false, false);
}

__device__ __forceinline__ v8f vzero8() {
  v8f z = {0.f, 0.f, 0.f, 0.f, 0.f, 0.f, 0.f, 0.f};
  return z;
}

// Build a 16-element bf16 fragment from two 16-byte LDS chunks (2x ds_load_b128).
__device__ __forceinline__ v16bf frag16(const __bf16* p0, const __bf16* p1) {
  v8bf lo = *(const v8bf*)p0;
  v8bf hi = *(const v8bf*)p1;
  v16bf r;
#pragma unroll
  for (int e = 0; e < 8; ++e) { r[e] = lo[e]; r[8 + e] = hi[e]; }
  return r;
}

// A-fragment (16x32, MxK): lane=hf*16+l16 holds row M=l16.
//   e in [0,16): K = hf*16 + e -> contiguous 32B
__device__ __forceinline__ v16bf a_frag(const __bf16* row, int hf) {
  return frag16(row + hf * 16, row + hf * 16 + 8);
}

// Contiguous 16-bf16 B-fragment (k = hf*16 + e).
__device__ __forceinline__ v16bf b_frag(const __bf16* p) {
  return frag16(p, p + 8);
}

__device__ __forceinline__ v8bf cvt8(const float4& a, const float4& b) {
  v8bf r;
  r[0] = (__bf16)a.x; r[1] = (__bf16)a.y; r[2] = (__bf16)a.z; r[3] = (__bf16)a.w;
  r[4] = (__bf16)b.x; r[5] = (__bf16)b.y; r[6] = (__bf16)b.z; r[7] = (__bf16)b.w;
  return r;
}

// ---------------------------------------------------------------------------
// Weight packing: B[K,N] f32 (row-major) -> Bt[N,K] bf16 (row-major).
// ---------------------------------------------------------------------------
__global__ void pack_bt_kernel(const float* __restrict__ B, __bf16* __restrict__ Bt,
                               int K, int N) {
  const int idx = blockIdx.x * 256 + threadIdx.x;  // = n*K + k
  if (idx >= N * K) return;
  const int n = idx / K, k = idx % K;
  Bt[idx] = (__bf16)B[(size_t)k * N + n];
}

// wo[n,d,h] -> Bt[d][k=(n*64+h)] bf16  (folds the 'benth,ndh->betd' permutation)
__global__ void pack_wo_bt_kernel(const float* __restrict__ wo, __bf16* __restrict__ Bt) {
  const int idx = blockIdx.x * 256 + threadIdx.x;  // = d*512 + k
  if (idx >= 512 * 512) return;
  const int d = idx >> 9, k = idx & 511;
  const int n = k >> 6, h = k & 63;
  Bt[idx] = (__bf16)wo[((size_t)n * 512 + d) * 64 + h];
}

// ---------------------------------------------------------------------------
// GEMM: C[M,N] = A[M,K](f32) * Bt[N,K](bf16 packed) + bias, optional ReLU.
// Block = 256 threads (8 waves); tile 256(M) x 64(N); K step 32.
// Double-buffered LDS; each wave computes 32x64 (8 WMMA per k-step).
// grid = (N/64, ceil(M/256)).
// ---------------------------------------------------------------------------
__global__ __launch_bounds__(256)
void gemm_bias_kernel(const float* __restrict__ A, const __bf16* __restrict__ Bt,
                      const float* __restrict__ bias, float* __restrict__ C,
                      int M, int K, int N, int relu) {
  __shared__ __align__(16) __bf16 Asm[2][256 * 32];  // 2 x 16KB
  __shared__ __align__(16) __bf16 Bsm[2][64 * 32];   // 2 x 4KB

  const int tid = threadIdx.x;
  const int lane = tid & 31, wave = tid >> 5;
  const int l16 = lane & 15, hf = lane >> 4;
  const int m_base = blockIdx.y * 256;
  const int n0 = blockIdx.x * 64;

  // staging roles: one full 32-float A row per thread; 8-bf16 B chunk per thread
  int arc = m_base + tid; if (arc >= M) arc = M - 1;
  const float* aptr = A + (size_t)arc * K;
  const int brow = tid >> 2, bseg = tid & 3;
  const __bf16* bptr = Bt + (size_t)(n0 + brow) * K + bseg * 8;

  float4 fa[8];
  v8bf breg;
  auto load_regs = [&](int k) {
    const float4* src = (const float4*)(aptr + k);
#pragma unroll
    for (int j = 0; j < 8; ++j) fa[j] = src[j];
    breg = *(const v8bf*)(bptr + k);
  };
  auto store_tile = [&](int buf) {
#pragma unroll
    for (int j = 0; j < 4; ++j)
      *(v8bf*)&Asm[buf][tid * 32 + j * 8] = cvt8(fa[2 * j], fa[2 * j + 1]);
    *(v8bf*)&Bsm[buf][brow * 32 + bseg * 8] = breg;
  };

  v8f acc[2][4];
#pragma unroll
  for (int i = 0; i < 2; ++i)
#pragma unroll
    for (int t = 0; t < 4; ++t) acc[i][t] = vzero8();

  const int m0w = wave * 32;

  // prologue: tile 0 into buffer 0
  load_regs(0);
  store_tile(0);

  int cur = 0;
  for (int k0 = 0; k0 < K; k0 += 32) {
    __syncthreads();  // tile(cur) visible; prev buffer free for overwrite
    const bool more = (k0 + 32) < K;
    if (more) {
      load_regs(k0 + 32);  // global b128 loads overlap with WMMA below
      __builtin_prefetch(aptr + k0 + 64, 0, 1);
      __builtin_prefetch(bptr + k0 + 64, 0, 1);
    }

    // ---- batched fragment loads (12x ds_load_b128), then 8 WMMA ----
    v16bf a0 = a_frag(&Asm[cur][(m0w + l16) * 32], hf);
    v16bf a1 = a_frag(&Asm[cur][(m0w + 16 + l16) * 32], hf);
    v16bf b0 = b_frag(&Bsm[cur][(0 * 16 + l16) * 32 + hf * 16]);
    v16bf b1 = b_frag(&Bsm[cur][(1 * 16 + l16) * 32 + hf * 16]);
    v16bf b2 = b_frag(&Bsm[cur][(2 * 16 + l16) * 32 + hf * 16]);
    v16bf b3 = b_frag(&Bsm[cur][(3 * 16 + l16) * 32 + hf * 16]);
    acc[0][0] = wmma_bf16(a0, b0, acc[0][0]);
    acc[1][0] = wmma_bf16(a1, b0, acc[1][0]);
    acc[0][1] = wmma_bf16(a0, b1, acc[0][1]);
    acc[1][1] = wmma_bf16(a1, b1, acc[1][1]);
    acc[0][2] = wmma_bf16(a0, b2, acc[0][2]);
    acc[1][2] = wmma_bf16(a1, b2, acc[1][2]);
    acc[0][3] = wmma_bf16(a0, b3, acc[0][3]);
    acc[1][3] = wmma_bf16(a1, b3, acc[1][3]);

    if (more) store_tile(cur ^ 1);  // protected by next iteration's barrier
    cur ^= 1;
  }

  const float bv0 = bias ? bias[n0 + l16 + 0]  : 0.f;
  const float bv1 = bias ? bias[n0 + l16 + 16] : 0.f;
  const float bv2 = bias ? bias[n0 + l16 + 32] : 0.f;
  const float bv3 = bias ? bias[n0 + l16 + 48] : 0.f;

#pragma unroll
  for (int i = 0; i < 2; ++i) {
#pragma unroll
    for (int r = 0; r < 8; ++r) {
      const int row = m_base + m0w + i * 16 + r + 8 * hf;
      if (row < M) {
        float* crow = C + (size_t)row * N + n0 + l16;
        float v0 = acc[i][0][r] + bv0, v1 = acc[i][1][r] + bv1;
        float v2 = acc[i][2][r] + bv2, v3 = acc[i][3][r] + bv3;
        if (relu) {
          v0 = fmaxf(v0, 0.f); v1 = fmaxf(v1, 0.f);
          v2 = fmaxf(v2, 0.f); v3 = fmaxf(v3, 0.f);
        }
        crow[0] = v0; crow[16] = v1; crow[32] = v2; crow[48] = v3;
      }
    }
  }
}

// ---------------------------------------------------------------------------
// Fused attention, one WG (8 waves) per (pair, head).  T=128, H=64, N=8.
// Q/K row-major and V transposed in LDS (bf16); P overlays dead Q+K region.
// Fragments batched before each WMMA group.
// ---------------------------------------------------------------------------
__global__ __launch_bounds__(256)
void attn_kernel(const float* __restrict__ Qg, const float* __restrict__ Kg,
                 const float* __restrict__ Vg, float* __restrict__ Og,
                 int qsel, int kvsel) {
  const int head = blockIdx.x;
  const int pair = blockIdx.y;
  const int qseq = qsel ? (pair & 31) : pair;
  const int kvseq = kvsel ? (pair >> 5) : pair;

  __shared__ __align__(16) __bf16 smem[128 * 128 + 64 * 128];  // 48KB
  __bf16* Qs = smem;                // [128][64]
  __bf16* Ks = smem + 128 * 64;     // [128][64]
  __bf16* Ps = smem;                // [128][128] (union over Qs+Ks)
  __bf16* Vt = smem + 128 * 128;    // [64][128]  (V transposed: Vt[h][t])

  const int tid = threadIdx.x;
  for (int i = tid; i < 128 * 64; i += 256) {
    const int t = i >> 6, h = i & 63;
    const size_t gq = (((size_t)qseq  * 128 + t) * 8 + head) * 64 + h;
    const size_t gk = (((size_t)kvseq * 128 + t) * 8 + head) * 64 + h;
    Qs[i] = (__bf16)Qg[gq];
    Ks[i] = (__bf16)Kg[gk];
    Vt[h * 128 + t] = (__bf16)Vg[gk];
  }
  __syncthreads();

  const int lane = tid & 31, wave = tid >> 5;
  const int l16 = lane & 15, hf = lane >> 4;
  const int m0 = wave * 16;  // this wave owns score rows [m0, m0+16)

  // ----- S = Q @ K^T * (1/sqrt(H)) : 8 tiles of 16x16 across 128 cols -----
  v8f s[8];
#pragma unroll
  for (int t = 0; t < 8; ++t) s[t] = vzero8();

  for (int k0 = 0; k0 < 64; k0 += 32) {
    v16bf a = a_frag(Qs + (size_t)(m0 + l16) * 64 + k0, hf);
    v16bf kb[8];
#pragma unroll
    for (int t = 0; t < 8; ++t)
      kb[t] = b_frag(Ks + (size_t)(t * 16 + l16) * 64 + k0 + hf * 16);
#pragma unroll
    for (int t = 0; t < 8; ++t) s[t] = wmma_bf16(a, kb[t], s[t]);
  }
  __syncthreads();  // all waves done with Qs/Ks before Ps overwrite

  // ----- row softmax (rows live across the 16-lane halves) -----
  const float scale = 0.125f;  // 1/sqrt(64)
#pragma unroll
  for (int r = 0; r < 8; ++r) {
    float mx = -3.4e38f;
#pragma unroll
    for (int t = 0; t < 8; ++t) { s[t][r] *= scale; mx = fmaxf(mx, s[t][r]); }
#pragma unroll
    for (int off = 1; off < 16; off <<= 1) mx = fmaxf(mx, __shfl_xor(mx, off, 32));
    float sum = 0.f;
#pragma unroll
    for (int t = 0; t < 8; ++t) { float p = __expf(s[t][r] - mx); s[t][r] = p; sum += p; }
#pragma unroll
    for (int off = 1; off < 16; off <<= 1) sum += __shfl_xor(sum, off, 32);
    const float inv = 1.f / sum;
    const int row = m0 + r + 8 * hf;
#pragma unroll
    for (int t = 0; t < 8; ++t)
      Ps[(size_t)row * 128 + t * 16 + l16] = (__bf16)(s[t][r] * inv);
  }
  __syncthreads();

  // ----- O = P @ V : 4 tiles of 16x16 across H=64 -----
  v8f o[4];
#pragma unroll
  for (int t = 0; t < 4; ++t) o[t] = vzero8();

  for (int k0 = 0; k0 < 128; k0 += 32) {
    v16bf a = a_frag(Ps + (size_t)(m0 + l16) * 128 + k0, hf);
    v16bf vb[4];
#pragma unroll
    for (int ht = 0; ht < 4; ++ht)
      vb[ht] = b_frag(Vt + (size_t)(ht * 16 + l16) * 128 + k0 + hf * 16);
#pragma unroll
    for (int ht = 0; ht < 4; ++ht) o[ht] = wmma_bf16(a, vb[ht], o[ht]);
  }

#pragma unroll
  for (int r = 0; r < 8; ++r) {
    const int row = m0 + r + 8 * hf;
#pragma unroll
    for (int ht = 0; ht < 4; ++ht)
      Og[(((size_t)pair * 128 + row) * 8 + head) * 64 + ht * 16 + l16] = o[ht][r];
  }
}

// ---------------------------------------------------------------------------
// y[tok] = LN( a[tok] + (has_res ? xres[rmap(tok)] : 0) ) * g + b  over D=512.
// One wave per token; block = (32,4).
// ---------------------------------------------------------------------------
__global__ __launch_bounds__(128)
void ln_add_kernel(const float* __restrict__ Aadd, const float* __restrict__ Xres,
                   const float* __restrict__ g, const float* __restrict__ b,
                   float* __restrict__ Y, int rsel, int has_res, int ntok) {
  const int tok = blockIdx.x * 4 + threadIdx.y;
  if (tok >= ntok) return;
  const int lane = threadIdx.x;
  const int seq = tok >> 7, t = tok & 127;
  const int rseq = rsel ? (seq & 31) : seq;
  const float* ap = Aadd + (size_t)tok * 512;
  const float* rp = Xres + ((size_t)rseq * 128 + t) * 512;

  float xv[16];
  float sum = 0.f, sq = 0.f;
#pragma unroll
  for (int j = 0; j < 16; ++j) {
    const int d = lane + 32 * j;
    float v = ap[d] + (has_res ? rp[d] : 0.f);
    xv[j] = v; sum += v; sq += v * v;
  }
#pragma unroll
  for (int off = 1; off < 32; off <<= 1) {
    sum += __shfl_xor(sum, off, 32);
    sq  += __shfl_xor(sq,  off, 32);
  }
  const float mean = sum * (1.f / 512.f);
  const float var = sq * (1.f / 512.f) - mean * mean;
  const float inv = rsqrtf(var + 1e-6f);
  float* yp = Y + (size_t)tok * 512;
#pragma unroll
  for (int j = 0; j < 16; ++j) {
    const int d = lane + 32 * j;
    yp[d] = (xv[j] - mean) * inv * g[d] + b[d];
  }
}

// P[s,d] = mean over tokens 1..127 of X[s,t,d]
__global__ void pool_kernel(const float* __restrict__ X, float* __restrict__ P, int nseq) {
  const int idx = blockIdx.x * 256 + threadIdx.x;
  if (idx >= nseq * 512) return;
  const int s = idx >> 9, d = idx & 511;
  float sum = 0.f;
  for (int t = 1; t < 128; ++t) sum += X[((size_t)s * 128 + t) * 512 + d];
  P[idx] = sum * (1.f / 127.f);
}

// out[p] = cosine(QF[p/32], CF[p]) with 1e-12 clamp, F=256. One wave per pair.
__global__ __launch_bounds__(32)
void dot_kernel(const float* __restrict__ QF, const float* __restrict__ CF,
                float* __restrict__ out) {
  const int p = blockIdx.x;
  const int qb = p >> 5;
  const int lane = threadIdx.x;
  float dq = 0.f, dc = 0.f, dd = 0.f;
#pragma unroll
  for (int j = 0; j < 8; ++j) {
    const int f = lane + 32 * j;
    const float a = QF[qb * 256 + f];
    const float c = CF[p * 256 + f];
    dq += a * a; dc += c * c; dd += a * c;
  }
#pragma unroll
  for (int off = 1; off < 32; off <<= 1) {
    dq += __shfl_xor(dq, off, 32);
    dc += __shfl_xor(dc, off, 32);
    dd += __shfl_xor(dd, off, 32);
  }
  if (lane == 0)
    out[p] = dd * rsqrtf(fmaxf(dq, 1e-12f)) * rsqrtf(fmaxf(dc, 1e-12f));
}

// ---------------------------------------------------------------------------
// Host orchestration
// ---------------------------------------------------------------------------
extern "C" void kernel_launch(void* const* d_in, const int* in_sizes, int n_in,
                              void* d_out, int out_size, void* d_ws, size_t ws_size,
                              hipStream_t stream) {
  (void)in_sizes; (void)n_in; (void)out_size; (void)ws_size;
  constexpr int D = 512, Nh = 8, H = 64, FF = 2048, F = 256, T = 128;

  const float* c_in = (const float*)d_in[0];
  const float* q_in = (const float*)d_in[1];
  const float* sa_w[4] = {(const float*)d_in[2], (const float*)d_in[4],
                          (const float*)d_in[6], (const float*)d_in[8]};   // wq wk wv wo
  const float* sa_b[4] = {(const float*)d_in[3], (const float*)d_in[5],
                          (const float*)d_in[7], (const float*)d_in[9]};   // bq bk bv bo
  const float* ca_w[4] = {(const float*)d_in[10], (const float*)d_in[12],
                          (const float*)d_in[14], (const float*)d_in[16]};
  const float* ca_b[4] = {(const float*)d_in[11], (const float*)d_in[13],
                          (const float*)d_in[15], (const float*)d_in[17]};
  const float* ln_g[3] = {(const float*)d_in[18], (const float*)d_in[20], (const float*)d_in[22]};
  const float* ln_b[3] = {(const float*)d_in[19], (const float*)d_in[21], (const float*)d_in[23]};
  const float* ffn_w1 = (const float*)d_in[24];
  const float* ffn_b1 = (const float*)d_in[25];
  const float* ffn_w2 = (const float*)d_in[26];
  const float* ffn_b2 = (const float*)d_in[27];
  const float* lnf_g = (const float*)d_in[28];
  const float* lnf_b = (const float*)d_in[29];
  const float* feat_wq = (const float*)d_in[30];
  const float* feat_bq = (const float*)d_in[31];
  const float* feat_wc = (const float*)d_in[32];
  const float* feat_bc = (const float*)d_in[33];

  // workspace layout (floats); HID aliases QB..OB (exactly 4 SEQ buffers)
  const size_t SEQ = (size_t)256 * T * D;  // 16,777,216 floats
  float* ws = (float*)d_ws;
  float* X0 = ws;
  float* X1 = X0 + SEQ;
  float* AO = X1 + SEQ;
  float* QB = AO + SEQ;
  float* KB = QB + SEQ;
  float* VB = KB + SEQ;
  float* OB = VB + SEQ;
  float* HID = QB;                     // 256*128*2048 = 4*SEQ, spans QB..OB
  float* CP = OB + SEQ;                // 256*512
  float* QP = CP + (size_t)256 * D;    // 8*512
  float* QF = QP + (size_t)8 * D;      // 8*256
  float* CF = QF + (size_t)8 * F;      // 256*256
  __bf16* BT = (__bf16*)(CF + (size_t)256 * F);  // up to 2048*512 bf16 (2MB)

  auto pack = [&](const float* B, int K, int N) {
    pack_bt_kernel<<<(N * K) / 256, 256, 0, stream>>>(B, BT, K, N);
  };
  auto gemm = [&](const float* A, const float* bias, float* C,
                  int M, int K, int N, int relu) {
    dim3 grid(N / 64, (M + 255) / 256);
    gemm_bias_kernel<<<grid, 256, 0, stream>>>(A, BT, bias, C, M, K, N, relu);
  };
  auto lnadd = [&](const float* a, const float* xres, const float* g, const float* b,
                   float* y, int rsel, int has_res, int ntok) {
    ln_add_kernel<<<(ntok + 3) / 4, dim3(32, 4), 0, stream>>>(a, xres, g, b, y,
                                                              rsel, has_res, ntok);
  };

  float* Xc = X0;
  float* Xn = X1;
  hipMemcpyAsync(Xc, c_in, (size_t)32 * T * D * sizeof(float),
                 hipMemcpyDeviceToDevice, stream);

  int S = 32;  // current number of x sequences
  for (int i = 0; i < 2; ++i) {
    const size_t woff = (size_t)i * D * Nh * H;  // == i*D*D
    const size_t boff = (size_t)i * Nh * H;      // == i*D

    // ---- self-attention ----
    pack(sa_w[0] + woff, D, D);  gemm(Xc, sa_b[0] + boff, QB, S * T, D, D, 0);
    pack(sa_w[1] + woff, D, D);  gemm(Xc, sa_b[1] + boff, KB, S * T, D, D, 0);
    pack(sa_w[2] + woff, D, D);  gemm(Xc, sa_b[2] + boff, VB, S * T, D, D, 0);
    attn_kernel<<<dim3(Nh, S), 256, 0, stream>>>(QB, KB, VB, OB, 0, 0);
    pack_wo_bt_kernel<<<(D * D) / 256, 256, 0, stream>>>(sa_w[3] + woff, BT);
    gemm(OB, sa_b[3] + (size_t)i * D, AO, S * T, D, D, 0);
    lnadd(AO, Xc, ln_g[0] + (size_t)i * D, ln_b[0] + (size_t)i * D, Xn, 0, 1, S * T);
    { float* t = Xc; Xc = Xn; Xn = t; }

    // ---- cross-attention (K/V from raw q, broadcast on layer 0) ----
    pack(ca_w[0] + woff, D, D);  gemm(Xc, ca_b[0] + boff, QB, S * T, D, D, 0);
    pack(ca_w[1] + woff, D, D);  gemm(q_in, ca_b[1] + boff, KB, 8 * T, D, D, 0);
    pack(ca_w[2] + woff, D, D);  gemm(q_in, ca_b[2] + boff, VB, 8 * T, D, D, 0);
    attn_kernel<<<dim3(Nh, 256), 256, 0, stream>>>(QB, KB, VB, OB,
                                                   (S == 32) ? 1 : 0, 1);
    pack_wo_bt_kernel<<<(D * D) / 256, 256, 0, stream>>>(ca_w[3] + woff, BT);
    gemm(OB, ca_b[3] + (size_t)i * D, AO, 256 * T, D, D, 0);
    lnadd(AO, Xc, ln_g[1] + (size_t)i * D, ln_b[1] + (size_t)i * D, Xn,
          (S == 32) ? 1 : 0, 1, 256 * T);
    { float* t = Xc; Xc = Xn; Xn = t; }
    S = 256;

    // ---- FFN ----
    pack(ffn_w1 + (size_t)i * D * FF, D, FF);
    gemm(Xc, ffn_b1 + (size_t)i * FF, HID, S * T, D, FF, 1);
    pack(ffn_w2 + (size_t)i * FF * D, FF, D);
    gemm(HID, ffn_b2 + (size_t)i * D, AO, S * T, FF, D, 0);
    lnadd(AO, Xc, ln_g[2] + (size_t)i * D, ln_b[2] + (size_t)i * D, Xn, 0, 1, S * T);
    { float* t = Xc; Xc = Xn; Xn = t; }
  }

  // final LN (no residual)
  lnadd(Xc, Xc, lnf_g, lnf_b, Xn, 0, 0, 256 * T);
  { float* t = Xc; Xc = Xn; Xn = t; }

  // pooled features, projection heads, cosine similarity
  pool_kernel<<<(256 * D) / 256, 256, 0, stream>>>(Xc, CP, 256);
  pool_kernel<<<(8 * D) / 256, 256, 0, stream>>>(q_in, QP, 8);
  pack(feat_wq, D, F);  gemm(QP, feat_bq, QF, 8, D, F, 0);
  pack(feat_wc, D, F);  gemm(CP, feat_bc, CF, 256, D, F, 0);
  dot_kernel<<<256, 32, 0, stream>>>(QF, CF, (float*)d_out);
}
